// GCN_42880953483607
// MI455X (gfx1250) — matrix operand order repaired and checked by reading, not compiled, not run on hardware
//
#include <hip/hip_runtime.h>

typedef __attribute__((ext_vector_type(16))) __bf16 v16bf;
typedef __attribute__((ext_vector_type(8)))  float  v8f;

#define HDIM 128

static __device__ __forceinline__ float4 ld4(const float* p) { return *(const float4*)p; }
static __device__ __forceinline__ float4 relu4(float4 v) {
    return make_float4(fmaxf(v.x, 0.f), fmaxf(v.y, 0.f), fmaxf(v.z, 0.f), fmaxf(v.w, 0.f));
}
static __device__ __forceinline__ void atomAdd(float* p, float v) { unsafeAtomicAdd(p, v); }

// ---- Convert W1/W2/W3 (128x128 f32, K-major) into transposed bf16 [n][k] ----
__global__ void cvt_weights_kernel(const float* __restrict__ W1, const float* __restrict__ W2,
                                   const float* __restrict__ W3, __bf16* __restrict__ Wt) {
    int idx = blockIdx.x * blockDim.x + threadIdx.x;      // 3*128*128 threads
    int which = idx >> 14;
    int rem = idx & 16383;
    int k = rem >> 7, n = rem & 127;
    const float* W = (which == 0) ? W1 : ((which == 1) ? W2 : W3);
    Wt[which * 16384 + n * HDIM + k] = (__bf16)W[k * HDIM + n];
}

// ---- Degree / norm ----
__global__ void deg_init_kernel(float* deg, int n) {
    int i = blockIdx.x * blockDim.x + threadIdx.x;
    if (i < n) deg[i] = 1.0f;                              // self-loop
}
__global__ void deg_edge_kernel(const long long* __restrict__ col, float* deg, int e) {
    int i = blockIdx.x * blockDim.x + threadIdx.x;
    if (i < e) atomAdd(&deg[(int)col[i]], 1.0f);
}
__global__ void deg_rsqrt_kernel(float* deg, int n) {
    int i = blockIdx.x * blockDim.x + threadIdx.x;
    if (i < n) deg[i] = rsqrtf(deg[i]);                    // deg >= 1 always
}

// ---- Precompute per-edge (row,col) as int32 + symmetric norm; cuts 3x re-read traffic ----
__global__ void edge_pre_kernel(const long long* __restrict__ erow, const long long* __restrict__ ecol,
                                const float* __restrict__ dis,
                                int2* __restrict__ erci, float* __restrict__ enrm, int e) {
    int i = blockIdx.x * blockDim.x + threadIdx.x;
    if (i >= e) return;
    int r = (int)erow[i], c = (int)ecol[i];
    erci[i] = make_int2(r, c);
    enrm[i] = dis[r] * dis[c];
}

// ---- Dense transform: out[N,128] = act(A)[N,128] @ Wt(bf16,[n][k]) + bias ----
// W staged in LDS once per block; per k-step all 8 B fragments are register-batched
// so the 8 WMMAs issue back-to-back instead of wait-0 per op.
template <bool RELU>
__global__ __launch_bounds__(256) void gemm_bias_kernel(const float* __restrict__ A,
                                                        const __bf16* __restrict__ Wt,
                                                        const float* __restrict__ bias,
                                                        float* __restrict__ out, int nrows) {
    __shared__ __bf16 sW[HDIM * HDIM];                     // 32 KB of 320 KB/WGP
    {
        const uint4* src = (const uint4*)Wt;
        uint4* dst = (uint4*)sW;
#pragma unroll
        for (int j = 0; j < 8; ++j)                        // 256 thr * 8 * 16B = 32 KB
            dst[threadIdx.x + j * 256] = src[threadIdx.x + j * 256];
    }
    __syncthreads();                                       // all waves reach barrier (guard below)

    const int wave = threadIdx.x >> 5;
    const int lane = threadIdx.x & 31;
    const int tile = blockIdx.x * 8 + wave;
    const int rowBase = tile * 16;
    if (rowBase >= nrows) return;                          // wave-uniform: EXEC stays all-1s
    const int hl = lane >> 4;                              // half-wave select
    const int ln = lane & 15;
    const float* arow = A + (size_t)(rowBase + ln) * HDIM;

    v8f acc[8];
#pragma unroll
    for (int nt = 0; nt < 8; ++nt) {
        float bv = bias[nt * 16 + ln];                     // C/D col = lane%16
        v8f c;
#pragma unroll
        for (int j = 0; j < 8; ++j) c[j] = bv;
        acc[nt] = c;
    }

#pragma unroll
    for (int ks = 0; ks < 4; ++ks) {
        // A fragment (contiguous k order chosen here, same as B): lane<16 -> K ko+[0..15]; lane>=16 -> K 16..31 of chunk
        const int ko = ks * 32 + hl * 16;
        float4 v0 = ld4(arow + ko),      v1 = ld4(arow + ko + 4);
        float4 v2 = ld4(arow + ko + 8),  v3 = ld4(arow + ko + 12);
        if (RELU) { v0 = relu4(v0); v1 = relu4(v1); v2 = relu4(v2); v3 = relu4(v3); }
        v16bf av;
        av[0]  = (__bf16)v0.x; av[1]  = (__bf16)v0.y; av[2]  = (__bf16)v0.z; av[3]  = (__bf16)v0.w;
        av[4]  = (__bf16)v1.x; av[5]  = (__bf16)v1.y; av[6]  = (__bf16)v1.z; av[7]  = (__bf16)v1.w;
        av[8]  = (__bf16)v2.x; av[9]  = (__bf16)v2.y; av[10] = (__bf16)v2.z; av[11] = (__bf16)v2.w;
        av[12] = (__bf16)v3.x; av[13] = (__bf16)v3.y; av[14] = (__bf16)v3.z; av[15] = (__bf16)v3.w;

        // Batch-load all 8 B fragments from LDS, then drain 8 WMMAs back-to-back.
        v16bf bf[8];
#pragma unroll
        for (int nt = 0; nt < 8; ++nt) {
            // B fragment, contiguous k order as for A: lanes 0-15 hold K 0..15, lanes 16-31 hold K 16..31 of this chunk.
            const __bf16* bp = sW + (size_t)(nt * 16 + ln) * HDIM + ks * 32 + hl * 16;
            bf[nt] = *(const v16bf*)bp;                    // two ds_load_b128
        }
#pragma unroll
        for (int nt = 0; nt < 8; ++nt) {
            acc[nt] = __builtin_amdgcn_wmma_f32_16x16x32_bf16(
                false, av, false, bf[nt], (short)0, acc[nt], false, false);
        }
    }

#pragma unroll
    for (int nt = 0; nt < 8; ++nt) {
#pragma unroll
        for (int r = 0; r < 8; ++r) {
            int m = r + hl * 8;                            // C/D layout: VGPR r -> row r / r+8
            out[(size_t)(rowBase + m) * HDIM + nt * 16 + ln] = acc[nt][r];
        }
    }
}

// ---- Aggregation: out[i] = dis[i]^2 * t[i]  (self-loop term; also initializes out, no zero-fill needed) ----
__global__ void agg_init_kernel(const float* __restrict__ t, const float* __restrict__ dis,
                                float* __restrict__ out, int n) {
    long long tid = (long long)blockIdx.x * blockDim.x + threadIdx.x;
    int i = (int)(tid >> 5), c = (int)(tid & 31);
    if (i >= n) return;
    float s = dis[i]; s *= s;
    float4 v = ld4(t + (size_t)i * HDIM + c * 4);
    *(float4*)(out + (size_t)i * HDIM + c * 4) = make_float4(v.x * s, v.y * s, v.z * s, v.w * s);
}

// ---- Edge scatter: out[col] += norm * t[row]  (L2-resident gathers + f32 atomics) ----
__global__ void agg_edge_kernel(const int2* __restrict__ erci, const float* __restrict__ enrm,
                                const float* __restrict__ t, float* __restrict__ out, int e) {
    long long tid = (long long)blockIdx.x * blockDim.x + threadIdx.x;
    int i = (int)(tid >> 5), c = (int)(tid & 31);
    if (i >= e) return;
    int2 rc = erci[i];
    float nrm = enrm[i];
    float4 v = ld4(t + (size_t)rc.x * HDIM + c * 4);
    float* o = out + (size_t)rc.y * HDIM + c * 4;
    atomAdd(o + 0, v.x * nrm); atomAdd(o + 1, v.y * nrm);
    atomAdd(o + 2, v.z * nrm); atomAdd(o + 3, v.w * nrm);
}

// ---- Pooling ----
__global__ void pool_init_kernel(float* pooled, float* cnt, int g) {
    int i = blockIdx.x * blockDim.x + threadIdx.x;
    if (i < g * HDIM) pooled[i] = 0.f;
    if (i < g) cnt[i] = 0.f;
}
__global__ void pool_accum_kernel(const float* __restrict__ h, const long long* __restrict__ batch,
                                  float* __restrict__ pooled, float* __restrict__ cnt, int n) {
    long long tid = (long long)blockIdx.x * blockDim.x + threadIdx.x;
    int i = (int)(tid >> 5), c = (int)(tid & 31);
    if (i >= n) return;
    int g = (int)batch[i];
    float4 v = relu4(ld4(h + (size_t)i * HDIM + c * 4));   // final ReLU fused here
    float* p = pooled + g * HDIM + c * 4;
    atomAdd(p + 0, v.x); atomAdd(p + 1, v.y); atomAdd(p + 2, v.z); atomAdd(p + 3, v.w);
    if (c == 0) atomAdd(&cnt[g], 1.0f);
}

// ---- Head: out[g] = dot(pooled[g], Wl)/max(cnt,1) + bl ; one wave32 per graph ----
__global__ void final_kernel(const float* __restrict__ pooled, const float* __restrict__ cnt,
                             const float* __restrict__ Wl, const float* __restrict__ bl,
                             float* __restrict__ out) {
    int g = blockIdx.x, lane = threadIdx.x;
    float s = 0.f;
#pragma unroll
    for (int j = 0; j < 4; ++j) s += pooled[g * HDIM + lane * 4 + j] * Wl[lane * 4 + j];
#pragma unroll
    for (int off = 16; off > 0; off >>= 1) s += __shfl_xor(s, off, 32);
    if (lane == 0) out[g] = s / fmaxf(cnt[g], 1.0f) + bl[0];
}

extern "C" void kernel_launch(void* const* d_in, const int* in_sizes, int n_in,
                              void* d_out, int out_size, void* d_ws, size_t ws_size,
                              hipStream_t stream) {
    const float*     x     = (const float*)d_in[0];
    const long long* eidx  = (const long long*)d_in[1];
    const long long* batch = (const long long*)d_in[2];
    // d_in[3] = dropout_rate (identity in eval mode)
    const float* W1 = (const float*)d_in[4];  const float* b1 = (const float*)d_in[5];
    const float* W2 = (const float*)d_in[6];  const float* b2 = (const float*)d_in[7];
    const float* W3 = (const float*)d_in[8];  const float* b3 = (const float*)d_in[9];
    const float* Wl = (const float*)d_in[10]; const float* bl = (const float*)d_in[11];

    const int N = in_sizes[0] / HDIM;
    const int E = in_sizes[1] / 2;
    const int G = out_size;
    const long long* erow = eidx;
    const long long* ecol = eidx + E;

    // Workspace carve-up (256B aligned)
    char* ws = (char*)d_ws;
    size_t off = 0;
    auto carve = [&](size_t bytes) -> void* {
        void* p = ws + off;
        off = (off + bytes + 255) & ~(size_t)255;
        return p;
    };
    float*  dis    = (float*)carve((size_t)N * 4);
    __bf16* Wt     = (__bf16*)carve(3 * 16384 * 2);
    float*  pooled = (float*)carve((size_t)G * HDIM * 4);
    float*  cnt    = (float*)carve((size_t)G * 4);
    int2*   erci   = (int2*)carve((size_t)E * 8);           // packed int32 (row,col)
    float*  enrm   = (float*)carve((size_t)E * 4);          // precomputed edge norm
    float*  B0     = (float*)carve((size_t)N * HDIM * 4);   // dense-transform result
    float*  B1     = (float*)carve((size_t)N * HDIM * 4);   // aggregate ping
    float*  B2     = (float*)carve((size_t)N * HDIM * 4);   // aggregate pong

    const int T = 256;
    const int gN   = (N + T - 1) / T;
    const int gE   = (E + T - 1) / T;
    const int gN32 = (int)(((long long)N * 32 + T - 1) / T);
    const int gE32 = (int)(((long long)E * 32 + T - 1) / T);
    const int gGemm = (N / 16 + 7) / 8;

    cvt_weights_kernel<<<3 * 16384 / T, T, 0, stream>>>(W1, W2, W3, Wt);
    deg_init_kernel<<<gN, T, 0, stream>>>(dis, N);
    deg_edge_kernel<<<gE, T, 0, stream>>>(ecol, dis, E);
    deg_rsqrt_kernel<<<gN, T, 0, stream>>>(dis, N);
    edge_pre_kernel<<<gE, T, 0, stream>>>(erow, ecol, dis, erci, enrm, E);

    // Layer 1: x -> B0 -> B1
    gemm_bias_kernel<false><<<gGemm, T, 0, stream>>>(x, Wt, b1, B0, N);
    agg_init_kernel<<<gN32, T, 0, stream>>>(B0, dis, B1, N);
    agg_edge_kernel<<<gE32, T, 0, stream>>>(erci, enrm, B0, B1, E);

    // Layer 2: relu(B1) -> B0 -> B2
    gemm_bias_kernel<true><<<gGemm, T, 0, stream>>>(B1, Wt + 16384, b2, B0, N);
    agg_init_kernel<<<gN32, T, 0, stream>>>(B0, dis, B2, N);
    agg_edge_kernel<<<gE32, T, 0, stream>>>(erci, enrm, B0, B2, E);

    // Layer 3: relu(B2) -> B0 -> B1
    gemm_bias_kernel<true><<<gGemm, T, 0, stream>>>(B2, Wt + 32768, b3, B0, N);
    agg_init_kernel<<<gN32, T, 0, stream>>>(B0, dis, B1, N);
    agg_edge_kernel<<<gE32, T, 0, stream>>>(erci, enrm, B0, B1, E);

    // Pool (ReLU fused) + head
    pool_init_kernel<<<(G * HDIM + T - 1) / T, T, 0, stream>>>(pooled, cnt, G);
    pool_accum_kernel<<<gN32, T, 0, stream>>>(B1, batch, pooled, cnt, N);
    final_kernel<<<G, 32, 0, stream>>>(pooled, cnt, Wl, bl, (float*)d_out);
}
